// Attention_72765335929035
// MI455X (gfx1250) — matrix operand order repaired and checked by reading, not compiled, not run on hardware
//
#include <hip/hip_runtime.h>
#include <hip/hip_bf16.h>

// ---------------------------------------------------------------------------
// Types for CDNA5 WMMA (wave32): v16bf A/B fragments, v8f accumulator.
// ---------------------------------------------------------------------------
typedef __bf16 bf16_t;
typedef __attribute__((ext_vector_type(16))) __bf16 v16bf;
typedef __attribute__((ext_vector_type(8)))  __bf16 v8bf;
typedef __attribute__((ext_vector_type(8)))  float  v8f;
typedef int v4i __attribute__((vector_size(16)));

union BFrag { v16bf v; v8bf h[2]; };

#define WMMA_BF16(A, B, C) \
  __builtin_amdgcn_wmma_f32_16x16x32_bf16(false, (A), false, (B), (short)0, (C), false, false)

// Problem constants
static constexpr int Bc = 2, Sc = 2048, Dc = 2048, Hc = 32, KVc = 8, HDc = 64;
static constexpr int NREPc = Hc / KVc;        // 4
static constexpr int Mrows = Bc * Sc;         // 4096

// ---------------------------------------------------------------------------
// Async copy to LDS (CDNA5 GLOBAL_LOAD_ASYNC_TO_LDS_B128, ASYNCcnt path).
// Probe-confirmed signature: (v4i addrspace(1)*, v4i addrspace(3)*, imm, imm).
// ---------------------------------------------------------------------------
#if defined(__has_builtin)
#if __has_builtin(__builtin_amdgcn_global_load_async_to_lds_b128)
#define HAVE_ASYNC_LDS 1
#endif
#endif
#ifndef HAVE_ASYNC_LDS
#define HAVE_ASYNC_LDS 0
#endif

#if HAVE_ASYNC_LDS
__device__ __forceinline__ void async_cp16(const bf16_t* g, bf16_t* l) {
  __builtin_amdgcn_global_load_async_to_lds_b128(
      (__attribute__((address_space(1))) v4i*)g,
      (__attribute__((address_space(3))) v4i*)l, 0, 0);
}
__device__ __forceinline__ void async_wait0() {
#if __has_builtin(__builtin_amdgcn_s_wait_asynccnt)
  __builtin_amdgcn_s_wait_asynccnt(0);
#else
  asm volatile("s_wait_asynccnt 0" ::: "memory");
#endif
}
#endif

// ---------------------------------------------------------------------------
// bf16 GEMM, both operands pre-packed bf16: A[M,K] row-major, Bt[N,K]
// row-major (i.e. B already transposed). 128x128 tile, BK=64, 256 threads,
// double-buffered LDS, async staging when available.
// ---------------------------------------------------------------------------
template <bool OUT_BF16>
__global__ __launch_bounds__(256) void gemm_bf16(
    const bf16_t* __restrict__ A, const bf16_t* __restrict__ Bt,
    void* __restrict__ Cout, int M, int N, int K)
{
  constexpr int BM = 128, BN = 128, BK = 64, KP = 72;  // 144B rows, 16B aligned
  __shared__ __align__(16) bf16_t As[2][BM * KP];
  __shared__ __align__(16) bf16_t Bs[2][BN * KP];

  const int tid  = threadIdx.x;
  const int lane = tid & 31;
  const int wave = tid >> 5;
  const int wr   = wave & 3;    // wave row 0..3  (32 rows each)
  const int wc   = wave >> 2;   // wave col 0..1  (64 cols each)
  const int half = lane >> 4;
  const int l16  = lane & 15;
  const int mBase = blockIdx.y * BM;
  const int nBase = blockIdx.x * BN;
  const int nSteps = K / BK;

  v8f acc[2][4];
  const v8f vzero = {0.f,0.f,0.f,0.f,0.f,0.f,0.f,0.f};
#pragma unroll
  for (int mt = 0; mt < 2; ++mt)
#pragma unroll
    for (int nt = 0; nt < 4; ++nt) acc[mt][nt] = vzero;

  // Staging: 1024 chunks of 8 bf16 per matrix (128 rows x 8 chunks); 4/thread.
#if HAVE_ASYNC_LDS
  auto stage_async = [&](int k0, int buf) {
#pragma unroll
    for (int i = 0; i < 4; ++i) {
      int c = tid + i * 256;
      int row = c >> 3, c8 = c & 7;
      async_cp16(A  + (size_t)(mBase + row) * K + k0 + c8 * 8,
                 &As[buf][row * KP + c8 * 8]);
      async_cp16(Bt + (size_t)(nBase + row) * K + k0 + c8 * 8,
                 &Bs[buf][row * KP + c8 * 8]);
    }
  };
#else
  v8bf ra[4], rb[4];
  auto load_tile = [&](int k0) {
#pragma unroll
    for (int i = 0; i < 4; ++i) {
      int c = tid + i * 256;
      int row = c >> 3, c8 = c & 7;
      ra[i] = *(const v8bf*)(A  + (size_t)(mBase + row) * K + k0 + c8 * 8);
      rb[i] = *(const v8bf*)(Bt + (size_t)(nBase + row) * K + k0 + c8 * 8);
    }
  };
  auto store_tile = [&](int buf) {
#pragma unroll
    for (int i = 0; i < 4; ++i) {
      int c = tid + i * 256;
      int row = c >> 3, c8 = c & 7;
      *(v8bf*)(&As[buf][row * KP + c8 * 8]) = ra[i];
      *(v8bf*)(&Bs[buf][row * KP + c8 * 8]) = rb[i];
    }
  };
#endif

  auto compute = [&](int buf) {
#pragma unroll
    for (int ks = 0; ks < 2; ++ks) {
      BFrag a[2], b[4];
#pragma unroll
      for (int mt = 0; mt < 2; ++mt) {
        const bf16_t* p = &As[buf][(wr * 32 + mt * 16 + l16) * KP + ks * 32];
        a[mt].h[0] = *(const v8bf*)(p + half * 8);        // K = half*8 .. +7
        a[mt].h[1] = *(const v8bf*)(p + 16 + half * 8);   // K = 16+half*8 .. +7
      }
#pragma unroll
      for (int nt = 0; nt < 4; ++nt) {
        const bf16_t* p = &Bs[buf][(wc * 64 + nt * 16 + l16) * KP + ks * 32];
        b[nt].h[0] = *(const v8bf*)(p + half * 8);        // K = half*8 .. +7
        b[nt].h[1] = *(const v8bf*)(p + 16 + half * 8);   // K = 16+half*8 .. +7
      }
#pragma unroll
      for (int mt = 0; mt < 2; ++mt)
#pragma unroll
        for (int nt = 0; nt < 4; ++nt)
          acc[mt][nt] = WMMA_BF16(a[mt].v, b[nt].v, acc[mt][nt]);
    }
  };

  // Software pipeline: prologue stage tile 0, one barrier per K-step.
#if HAVE_ASYNC_LDS
  stage_async(0, 0);
  async_wait0();
  __syncthreads();
  for (int j = 0; j < nSteps; ++j) {
    int cur = j & 1;
    if (j + 1 < nSteps) stage_async((j + 1) * BK, 1 - cur);
    compute(cur);
    if (j + 1 < nSteps) async_wait0();
    __syncthreads();
  }
#else
  load_tile(0);
  store_tile(0);
  __syncthreads();
  for (int j = 0; j < nSteps; ++j) {
    int cur = j & 1;
    if (j + 1 < nSteps) load_tile((j + 1) * BK);
    compute(cur);
    if (j + 1 < nSteps) store_tile(1 - cur);
    __syncthreads();
  }
#endif

  // Epilogue: C layout row = v + 8*half, col = l16.
#pragma unroll
  for (int mt = 0; mt < 2; ++mt)
#pragma unroll
    for (int nt = 0; nt < 4; ++nt) {
      int col = nBase + wc * 64 + nt * 16 + l16;
#pragma unroll
      for (int v = 0; v < 8; ++v) {
        int row = mBase + wr * 32 + mt * 16 + v + 8 * half;
        if (OUT_BF16)
          ((bf16_t*)Cout)[(size_t)row * N + col] = (bf16_t)acc[mt][nt][v];
        else
          ((float*)Cout)[(size_t)row * N + col] = acc[mt][nt][v];
      }
    }
}

// ---------------------------------------------------------------------------
// f32 -> bf16 elementwise convert (8 elems/thread).
// ---------------------------------------------------------------------------
__global__ __launch_bounds__(256) void cvt_bf16(const float* __restrict__ in,
                                                bf16_t* __restrict__ out, long n) {
  long i = ((long)blockIdx.x * 256 + threadIdx.x) * 8;
  if (i + 8 <= n) {
    float4 a = *(const float4*)(in + i);
    float4 b = *(const float4*)(in + i + 4);
    v8bf o;
    o[0] = (bf16_t)a.x; o[1] = (bf16_t)a.y; o[2] = (bf16_t)a.z; o[3] = (bf16_t)a.w;
    o[4] = (bf16_t)b.x; o[5] = (bf16_t)b.y; o[6] = (bf16_t)b.z; o[7] = (bf16_t)b.w;
    *(v8bf*)(out + i) = o;
  }
}

// ---------------------------------------------------------------------------
// Transpose + convert: in f32 (R x C) -> out bf16 (C x R). 32x32 LDS tile.
// ---------------------------------------------------------------------------
__global__ __launch_bounds__(256) void transpose_cvt(
    const float* __restrict__ in, bf16_t* __restrict__ out, int R, int C) {
  __shared__ float tile[32][33];
  const int c0 = blockIdx.x * 32;
  const int r0 = blockIdx.y * 32;
  const int tx = threadIdx.x, ty = threadIdx.y;  // block (32, 8)
#pragma unroll
  for (int i = ty; i < 32; i += 8)
    tile[i][tx] = in[(size_t)(r0 + i) * C + c0 + tx];
  __syncthreads();
#pragma unroll
  for (int i = ty; i < 32; i += 8)
    out[(size_t)(c0 + i) * R + r0 + tx] = (bf16_t)tile[tx][i];
}

// ---------------------------------------------------------------------------
// RoPE + pack (bf16 in, bf16 out): Q(b,h,s,hd)*0.125, K(b,kv,s,hd),
// V transposed (b,kv,hd,s).
// ---------------------------------------------------------------------------
__global__ __launch_bounds__(256) void rope_pack(
    const bf16_t* __restrict__ xq, const bf16_t* __restrict__ xk,
    const bf16_t* __restrict__ xv, const float* __restrict__ fcos,
    const float* __restrict__ fsin, bf16_t* __restrict__ Qb,
    bf16_t* __restrict__ Kb, bf16_t* __restrict__ Vt)
{
  constexpr int HP = HDc / 2;                         // 32 rotation pairs
  const long NQ = (long)Bc * Sc * Hc * HP;            // 4194304
  const long NK = (long)Bc * Sc * KVc * HP;           // 1048576
  long i = (long)blockIdx.x * 256 + threadIdx.x;

  if (i < NQ) {
    int hp = (int)(i & 31); long t = i >> 5;
    int h = (int)(t % Hc); t /= Hc;
    int s = (int)(t % Sc); int b = (int)(t / Sc);
    const bf16_t* p = xq + (((size_t)(b * Sc + s) * Hc) + h) * HDc + 2 * hp;
    float re = (float)p[0], im = (float)p[1];
    float c = fcos[s * HP + hp], sn = fsin[s * HP + hp];
    bf16_t* o = Qb + ((size_t)(b * Hc + h) * Sc + s) * HDc + 2 * hp;
    o[0] = (bf16_t)(0.125f * (re * c - im * sn));     // fold 1/sqrt(HD)
    o[1] = (bf16_t)(0.125f * (re * sn + im * c));
  } else if (i < NQ + NK) {
    long j = i - NQ;
    int hp = (int)(j & 31); long t = j >> 5;
    int kv = (int)(t % KVc); t /= KVc;
    int s = (int)(t % Sc); int b = (int)(t / Sc);
    const bf16_t* p = xk + (((size_t)(b * Sc + s) * KVc) + kv) * HDc + 2 * hp;
    float re = (float)p[0], im = (float)p[1];
    float c = fcos[s * HP + hp], sn = fsin[s * HP + hp];
    bf16_t* o = Kb + ((size_t)(b * KVc + kv) * Sc + s) * HDc + 2 * hp;
    o[0] = (bf16_t)(re * c - im * sn);
    o[1] = (bf16_t)(re * sn + im * c);
  } else if (i < NQ + 2 * NK) {
    long j = i - NQ - NK;
    int hp = (int)(j & 31); long t = j >> 5;
    int kv = (int)(t % KVc); t /= KVc;
    int s = (int)(t % Sc); int b = (int)(t / Sc);
    const bf16_t* p = xv + (((size_t)(b * Sc + s) * KVc) + kv) * HDc + 2 * hp;
    bf16_t* o = Vt + ((size_t)(b * KVc + kv) * HDc) * Sc;
    o[(size_t)(2 * hp + 0) * Sc + s] = p[0];
    o[(size_t)(2 * hp + 1) * Sc + s] = p[1];
  }
}

// ---------------------------------------------------------------------------
// Flash attention, causal, GQA. Grid: (S/64, B*H). 4 waves x 16 q-rows.
// 64-key blocks, double-buffered K/V LDS tiles, async staging when available.
// ---------------------------------------------------------------------------
__global__ __launch_bounds__(128) void flash_attn(
    const bf16_t* __restrict__ Qb, const bf16_t* __restrict__ Kb,
    const bf16_t* __restrict__ Vt, bf16_t* __restrict__ Ob)
{
  constexpr int KP = 72;                           // padded row of 64 bf16
  __shared__ __align__(16) bf16_t Kls[2][64 * KP]; // (key, hd)
  __shared__ __align__(16) bf16_t Vls[2][64 * KP]; // (hd, key)  (V^T)
  __shared__ __align__(16) bf16_t Pls[4 * 16 * 64];

  const int tid  = threadIdx.x;
  const int lane = tid & 31;
  const int wave = tid >> 5;
  const int half = lane >> 4;
  const int l16  = lane & 15;
  const int head = blockIdx.y;
  const int b = head / Hc, h = head % Hc, kv = h / NREPc;
  const int qBase = blockIdx.x * 64;
  const int qRow0 = qBase + wave * 16;

  const bf16_t* Qh = Qb + (size_t)(b * Hc + h) * Sc * HDc;
  const bf16_t* Kh = Kb + (size_t)(b * KVc + kv) * Sc * HDc;
  const bf16_t* Vh = Vt + (size_t)(b * KVc + kv) * HDc * Sc;

  // Q A-fragments (2 k-slices of 32 over HD=64), kept in registers.
  BFrag aq[2];
  {
    const bf16_t* qp = Qh + (size_t)(qRow0 + l16) * HDc;
    aq[0].h[0] = *(const v8bf*)(qp + half * 8);
    aq[0].h[1] = *(const v8bf*)(qp + 16 + half * 8);
    aq[1].h[0] = *(const v8bf*)(qp + 32 + half * 8);
    aq[1].h[1] = *(const v8bf*)(qp + 48 + half * 8);
  }

  const v8f vzero = {0.f,0.f,0.f,0.f,0.f,0.f,0.f,0.f};
  v8f o[4] = {vzero, vzero, vzero, vzero};
  float m[8], l[8];
#pragma unroll
  for (int v = 0; v < 8; ++v) { m[v] = -__builtin_inff(); l[v] = 0.f; }

  // Staging: K tile 64x64 and V^T tile 64x64, 512 chunks of 8 each; 4/thread.
#if HAVE_ASYNC_LDS
  auto stage_async = [&](int kb, int buf) {
#pragma unroll
    for (int i = 0; i < 4; ++i) {
      int c = tid + i * 128;
      int r = c >> 3, c8 = c & 7;
      async_cp16(Kh + (size_t)(kb + r) * HDc + c8 * 8, &Kls[buf][r * KP + c8 * 8]);
      async_cp16(Vh + (size_t)r * Sc + kb + c8 * 8,    &Vls[buf][r * KP + c8 * 8]);
    }
  };
#else
  v8bf rk[4], rv[4];
  auto load_tile = [&](int kb) {
#pragma unroll
    for (int i = 0; i < 4; ++i) {
      int c = tid + i * 128;
      int r = c >> 3, c8 = c & 7;
      rk[i] = *(const v8bf*)(Kh + (size_t)(kb + r) * HDc + c8 * 8);
      rv[i] = *(const v8bf*)(Vh + (size_t)r * Sc + kb + c8 * 8);
    }
  };
  auto store_tile = [&](int buf) {
#pragma unroll
    for (int i = 0; i < 4; ++i) {
      int c = tid + i * 128;
      int r = c >> 3, c8 = c & 7;
      *(v8bf*)(&Kls[buf][r * KP + c8 * 8]) = rk[i];
      *(v8bf*)(&Vls[buf][r * KP + c8 * 8]) = rv[i];
    }
  };
#endif

  auto compute = [&](int kb, int buf) {
    // scores: S = Q * K^T   (four 16-key subtiles, HD=64 in 2 k-slices)
    v8f sc[4];
#pragma unroll
    for (int nt = 0; nt < 4; ++nt) {
      v8f s = vzero;
      const int key = nt * 16 + l16;
#pragma unroll
      for (int ks = 0; ks < 2; ++ks) {
        BFrag bk;
        const bf16_t* p = &Kls[buf][key * KP + ks * 32];
        bk.h[0] = *(const v8bf*)(p + half * 8);
        bk.h[1] = *(const v8bf*)(p + 16 + half * 8);
        s = WMMA_BF16(aq[ks].v, bk.v, s);
      }
      sc[nt] = s;
    }

    // causal mask + row max (row r = v + 8*half across 16 lanes)
    float mt_[8];
#pragma unroll
    for (int v = 0; v < 8; ++v) {
      int q = qRow0 + v + 8 * half;
      float mv = -__builtin_inff();
#pragma unroll
      for (int nt = 0; nt < 4; ++nt) {
        int key = kb + nt * 16 + l16;
        if (key > q) sc[nt][v] = -__builtin_inff();
        mv = fmaxf(mv, sc[nt][v]);
      }
#pragma unroll
      for (int off = 8; off >= 1; off >>= 1)
        mv = fmaxf(mv, __shfl_xor(mv, off, 32));
      mt_[v] = mv;
    }
    // online softmax update
#pragma unroll
    for (int v = 0; v < 8; ++v) {
      float mNew = fmaxf(m[v], mt_[v]);
      float scale = __expf(m[v] - mNew);
      float rs = 0.f;
#pragma unroll
      for (int nt = 0; nt < 4; ++nt) {
        float p = __expf(sc[nt][v] - mNew);
        sc[nt][v] = p;
        rs += p;
      }
#pragma unroll
      for (int off = 8; off >= 1; off >>= 1)
        rs += __shfl_xor(rs, off, 32);
      l[v] = l[v] * scale + rs;
      m[v] = mNew;
#pragma unroll
      for (int ht = 0; ht < 4; ++ht) o[ht][v] *= scale;
    }

    // P (16x64) through per-wave LDS -> A-fragments (in-wave DS is in-order)
    bf16_t* P = Pls + wave * 1024;
#pragma unroll
    for (int v = 0; v < 8; ++v) {
      int r = v + 8 * half;
#pragma unroll
      for (int nt = 0; nt < 4; ++nt)
        P[r * 64 + nt * 16 + l16] = (bf16_t)sc[nt][v];
    }
    BFrag ap[2];
#pragma unroll
    for (int ks = 0; ks < 2; ++ks) {
      const bf16_t* p = P + l16 * 64 + ks * 32;
      ap[ks].h[0] = *(const v8bf*)(p + half * 8);
      ap[ks].h[1] = *(const v8bf*)(p + 16 + half * 8);
    }
    // O += P * V   (4 hd subtiles x 2 key slices)
#pragma unroll
    for (int ht = 0; ht < 4; ++ht) {
#pragma unroll
      for (int ks = 0; ks < 2; ++ks) {
        BFrag bv;
        const bf16_t* p = &Vls[buf][(ht * 16 + l16) * KP + ks * 32];
        bv.h[0] = *(const v8bf*)(p + half * 8);
        bv.h[1] = *(const v8bf*)(p + 16 + half * 8);
        o[ht] = WMMA_BF16(ap[ks].v, bv.v, o[ht]);
      }
    }
  };

  const int nkb = blockIdx.x + 1;  // causal: key blocks of 64 up to qBase+63
#if HAVE_ASYNC_LDS
  stage_async(0, 0);
  async_wait0();
  __syncthreads();
  for (int j = 0; j < nkb; ++j) {
    int cur = j & 1;
    if (j + 1 < nkb) stage_async((j + 1) * 64, 1 - cur);
    compute(j * 64, cur);
    if (j + 1 < nkb) async_wait0();
    __syncthreads();
  }
#else
  load_tile(0);
  store_tile(0);
  __syncthreads();
  for (int j = 0; j < nkb; ++j) {
    int cur = j & 1;
    if (j + 1 < nkb) load_tile((j + 1) * 64);
    compute(j * 64, cur);
    if (j + 1 < nkb) store_tile(1 - cur);
    __syncthreads();
  }
#endif

  // normalize and write attention output as bf16 in (b, s, h*hd)
#pragma unroll
  for (int v = 0; v < 8; ++v) {
    float rl = 1.0f / l[v];
    int q = qRow0 + v + 8 * half;
    bf16_t* op = Ob + ((size_t)(b * Sc + q) * Hc + h) * HDc;
#pragma unroll
    for (int ht = 0; ht < 4; ++ht)
      op[ht * 16 + l16] = (bf16_t)(o[ht][v] * rl);
  }
}

// ---------------------------------------------------------------------------
// Host-side launch
// ---------------------------------------------------------------------------
extern "C" void kernel_launch(void* const* d_in, const int* in_sizes, int n_in,
                              void* d_out, int out_size, void* d_ws, size_t ws_size,
                              hipStream_t stream) {
  const float* x    = (const float*)d_in[0];
  const float* wq   = (const float*)d_in[1];
  const float* wk   = (const float*)d_in[2];
  const float* wv   = (const float*)d_in[3];
  const float* wo   = (const float*)d_in[4];
  const float* fcos = (const float*)d_in[5];
  const float* fsin = (const float*)d_in[6];

  size_t off = 0;
  auto alloc = [&](size_t bytes) -> void* {
    void* p = (char*)d_ws + off;
    off += (bytes + 255) & ~(size_t)255;
    return p;
  };
  bf16_t* xb  = (bf16_t*)alloc((size_t)Mrows * Dc * 2);
  bf16_t* wqT = (bf16_t*)alloc((size_t)2048 * Dc * 2);
  bf16_t* wkT = (bf16_t*)alloc((size_t)512 * Dc * 2);
  bf16_t* wvT = (bf16_t*)alloc((size_t)512 * Dc * 2);
  bf16_t* woT = (bf16_t*)alloc((size_t)2048 * Dc * 2);
  bf16_t* xq  = (bf16_t*)alloc((size_t)Mrows * 2048 * 2);
  bf16_t* xk  = (bf16_t*)alloc((size_t)Mrows * 512 * 2);
  bf16_t* xv  = (bf16_t*)alloc((size_t)Mrows * 512 * 2);
  bf16_t* Qb  = (bf16_t*)alloc((size_t)Mrows * 2048 * 2);
  bf16_t* Kb  = (bf16_t*)alloc((size_t)Mrows * 512 * 2);
  bf16_t* Vt  = (bf16_t*)alloc((size_t)Mrows * 512 * 2);
  bf16_t* At  = (bf16_t*)alloc((size_t)Mrows * 2048 * 2);

  // pre-pass: x -> bf16; weights -> bf16 transposed (N x K)
  cvt_bf16<<<(Mrows * Dc) / (256 * 8), 256, 0, stream>>>(x, xb, (long)Mrows * Dc);
  transpose_cvt<<<dim3(64, 64), dim3(32, 8), 0, stream>>>(wq, wqT, 2048, 2048);
  transpose_cvt<<<dim3(16, 64), dim3(32, 8), 0, stream>>>(wk, wkT, 2048, 512);
  transpose_cvt<<<dim3(16, 64), dim3(32, 8), 0, stream>>>(wv, wvT, 2048, 512);
  transpose_cvt<<<dim3(64, 64), dim3(32, 8), 0, stream>>>(wo, woT, 2048, 2048);

  // QKV projections (bf16 out)
  gemm_bf16<true><<<dim3(2048 / 128, Mrows / 128), 256, 0, stream>>>(
      xb, wqT, xq, Mrows, 2048, Dc);
  gemm_bf16<true><<<dim3(512 / 128, Mrows / 128), 256, 0, stream>>>(
      xb, wkT, xk, Mrows, 512, Dc);
  gemm_bf16<true><<<dim3(512 / 128, Mrows / 128), 256, 0, stream>>>(
      xb, wvT, xv, Mrows, 512, Dc);

  // RoPE + pack (Q scaled, V transposed)
  {
    long total = (long)Bc * Sc * Hc * 32 + 2L * Bc * Sc * KVc * 32;  // 6291456
    rope_pack<<<(int)(total / 256), 256, 0, stream>>>(xq, xk, xv, fcos, fsin,
                                                      Qb, Kb, Vt);
  }

  // causal flash attention
  flash_attn<<<dim3(Sc / 64, Bc * Hc), 128, 0, stream>>>(Qb, Kb, Vt, At);

  // output projection -> fp32 d_out
  gemm_bf16<false><<<dim3(2048 / 128, Mrows / 128), 256, 0, stream>>>(
      At, woT, d_out, Mrows, 2048, Dc);
}
